// RelativeMultiHeadedAttention_84035330114266
// MI455X (gfx1250) — compile-verified, matrix operand order checked by reading, not run on hardware
//
#include <hip/hip_runtime.h>
#include <hip/hip_bf16.h>
#include <stdint.h>

// ---------------------------------------------------------------------------
// Transformer-XL relative multi-head attention, fused for MI455X (gfx1250).
//   A = (q+cb) @ k^T + shift((q+pb) @ Q)        (shift resolved elementwise)
// All GEMMs on v_wmma_f32_16x16x32_bf16 (wave32). Flash-style online softmax
// so the (B,H,L,L) score/S tensors are never materialized in HBM.
// Round 2: double-buffered GEMM K-loop (overlap loads with WMMA), LDS-shared
// k/v tiles in the attention kernel, global_prefetch of the next key block.
// ---------------------------------------------------------------------------

typedef __bf16 bf16;
typedef __attribute__((ext_vector_type(16))) __bf16 v16bf;
typedef __attribute__((ext_vector_type(8)))  float  v8f;

#define Bn   2
#define Ln   2048
#define Dn   512
#define Hn   8
#define DKn  64
#define BLn  (Bn * Ln)      // 4096 rows of E / Ev / O / y
#define LOG2E 1.4426950408889634f

union FragBF { v16bf v; uint4 q[2]; };

// A-operand fragment (16x32 bf16) from row-major [rows x ld] memory.
// lane&15 -> row; half = lane>>4 selects K chunks {half*8..+8, 16+half*8..+8}.
// B operands are loaded with the same routine from B^T stored row-major,
// i.e. every WMMA computes C = A * B^T.
__device__ __forceinline__ v16bf frag_ld(const bf16* __restrict__ p, int ld,
                                         int row0, int k0, int lane) {
  int r  = row0 + (lane & 15);
  int kc = k0 + ((lane >> 4) << 3);
  FragBF f;
  f.q[0] = *(const uint4*)(p + (size_t)r * ld + kc);
  f.q[1] = *(const uint4*)(p + (size_t)r * ld + kc + 16);
  return f.v;
}

__device__ __forceinline__ v16bf frag_ld_clamp(const bf16* __restrict__ p, int ld,
                                               int row0, int k0, int lane, int rmax) {
  int r = row0 + (lane & 15);
  r = r < 0 ? 0 : (r > rmax ? rmax : r);
  int kc = k0 + ((lane >> 4) << 3);
  FragBF f;
  f.q[0] = *(const uint4*)(p + (size_t)r * ld + kc);
  f.q[1] = *(const uint4*)(p + (size_t)r * ld + kc + 16);
  return f.v;
}

__device__ __forceinline__ v8f zero8() {
  v8f z; for (int e = 0; e < 8; ++e) z[e] = 0.f; return z;
}

#define WMMA_BF16(a, b, c) \
  __builtin_amdgcn_wmma_f32_16x16x32_bf16(false, (a), false, (b), (short)0, (c), false, false)

// 16x64 output tile per wave (4 D-fragments). Double-buffered K loop: the
// next step's operand loads are issued before the current step's WMMAs so the
// load wait overlaps matrix math instead of serializing with it.
__device__ __forceinline__ void gemm_tile(const bf16* __restrict__ A, int lda,
                                          const bf16* __restrict__ Bt, int ldb,
                                          int row0, int col0, int K, int lane,
                                          v8f out[4]) {
  for (int n = 0; n < 4; ++n) out[n] = zero8();
  v16bf a  = frag_ld(A, lda, row0, 0, lane);
  v16bf b0 = frag_ld(Bt, ldb, col0,      0, lane);
  v16bf b1 = frag_ld(Bt, ldb, col0 + 16, 0, lane);
  v16bf b2 = frag_ld(Bt, ldb, col0 + 32, 0, lane);
  v16bf b3 = frag_ld(Bt, ldb, col0 + 48, 0, lane);
  for (int k0 = 0; k0 < K; k0 += 32) {
    int kn = (k0 + 32 < K) ? k0 + 32 : 0;   // last iter reloads k0=0 (discarded)
    v16bf an = frag_ld(A, lda, row0, kn, lane);
    v16bf c0 = frag_ld(Bt, ldb, col0,      kn, lane);
    v16bf c1 = frag_ld(Bt, ldb, col0 + 16, kn, lane);
    v16bf c2 = frag_ld(Bt, ldb, col0 + 32, kn, lane);
    v16bf c3 = frag_ld(Bt, ldb, col0 + 48, kn, lane);
    out[0] = WMMA_BF16(a, b0, out[0]);
    out[1] = WMMA_BF16(a, b1, out[1]);
    out[2] = WMMA_BF16(a, b2, out[2]);
    out[3] = WMMA_BF16(a, b3, out[3]);
    a = an; b0 = c0; b1 = c1; b2 = c2; b3 = c3;
  }
}

// --------------------------- prep kernels ----------------------------------
__global__ void k_f2bf(const float* __restrict__ x, bf16* __restrict__ y, int n) {
  int i = blockIdx.x * blockDim.x + threadIdx.x;
  if (i < n) y[i] = (bf16)x[i];
}

// y[c*R + r] = x[r*C + c]   (transpose + convert)
__global__ void k_transpose_bf(const float* __restrict__ x, bf16* __restrict__ y,
                               int R, int C) {
  int i = blockIdx.x * blockDim.x + threadIdx.x;
  if (i < R * C) {
    int r = i / C, c = i % C;
    y[(size_t)c * R + r] = (bf16)x[i];
  }
}

// --------------------------- projection GEMMs ------------------------------
// grid: (N/64, M/64), 128 threads (4 waves); wave w owns rows blockIdx.y*64+w*16.

__global__ __launch_bounds__(128)
void k_proj_q(const bf16* __restrict__ E, const bf16* __restrict__ Wt,
              const float* __restrict__ cb, const float* __restrict__ pb,
              bf16* __restrict__ oqk, bf16* __restrict__ oqr) {
  int lane = threadIdx.x & 31, w = threadIdx.x >> 5;
  int row0 = (blockIdx.y << 6) + (w << 4), col0 = blockIdx.x << 6;
  v8f out[4];
  gemm_tile(E, Dn, Wt, Dn, row0, col0, Dn, lane, out);
  int half8 = ((lane >> 4) << 3), ln15 = lane & 15;
  for (int d = 0; d < 4; ++d)
    for (int e = 0; e < 8; ++e) {
      int m = row0 + e + half8, n = col0 + (d << 4) + ln15;
      int b = m >> 11, l = m & (Ln - 1), h = n >> 6, dk = n & 63;
      size_t idx = ((size_t)((b * Hn + h) * Ln + l)) * DKn + dk;
      float v = out[d][e];
      oqk[idx] = (bf16)(v + cb[n]);   // scores vs keys use q + cb
      oqr[idx] = (bf16)(v + pb[n]);   // scores vs positions use q + pb
    }
}

__global__ __launch_bounds__(128)
void k_proj_k(const bf16* __restrict__ E, const bf16* __restrict__ Wt,
              bf16* __restrict__ ok) {
  int lane = threadIdx.x & 31, w = threadIdx.x >> 5;
  int row0 = (blockIdx.y << 6) + (w << 4), col0 = blockIdx.x << 6;
  v8f out[4];
  gemm_tile(E, Dn, Wt, Dn, row0, col0, Dn, lane, out);
  int half8 = ((lane >> 4) << 3), ln15 = lane & 15;
  for (int d = 0; d < 4; ++d)
    for (int e = 0; e < 8; ++e) {
      int m = row0 + e + half8, n = col0 + (d << 4) + ln15;
      int b = m >> 11, l = m & (Ln - 1), h = n >> 6, dk = n & 63;
      ok[((size_t)((b * Hn + h) * Ln + l)) * DKn + dk] = (bf16)out[d][e];
    }
}

__global__ __launch_bounds__(128)
void k_proj_v(const bf16* __restrict__ Ev, const bf16* __restrict__ Wt,
              bf16* __restrict__ ovt) {   // transposed store: (B,H,DK,L)
  int lane = threadIdx.x & 31, w = threadIdx.x >> 5;
  int row0 = (blockIdx.y << 6) + (w << 4), col0 = blockIdx.x << 6;
  v8f out[4];
  gemm_tile(Ev, Dn, Wt, Dn, row0, col0, Dn, lane, out);
  int half8 = ((lane >> 4) << 3), ln15 = lane & 15;
  for (int d = 0; d < 4; ++d)
    for (int e = 0; e < 8; ++e) {
      int m = row0 + e + half8, n = col0 + (d << 4) + ln15;
      int b = m >> 11, l = m & (Ln - 1), h = n >> 6, dk = n & 63;
      ovt[((size_t)((b * Hn + h) * DKn + dk)) * Ln + l] = (bf16)out[d][e];
    }
}

// Qm[h, m, dk] = (Wkr @ R)[h*64+dk, m] = RT[m,:] . Wkr[h*64+dk,:]
__global__ __launch_bounds__(128)
void k_qm(const bf16* __restrict__ RT, const bf16* __restrict__ Wkr,
          bf16* __restrict__ qm) {
  int lane = threadIdx.x & 31, w = threadIdx.x >> 5;
  int row0 = (blockIdx.y << 6) + (w << 4), col0 = blockIdx.x << 6;
  v8f out[4];
  gemm_tile(RT, Dn, Wkr, Dn, row0, col0, Dn, lane, out);
  int half8 = ((lane >> 4) << 3), ln15 = lane & 15;
  for (int d = 0; d < 4; ++d)
    for (int e = 0; e < 8; ++e) {
      int m = row0 + e + half8, n = col0 + (d << 4) + ln15;
      int h = n >> 6, dk = n & 63;
      qm[((size_t)(h * Ln + m)) * DKn + dk] = (bf16)out[d][e];
    }
}

// y = O @ Wo^T + Wo_b + Ev  (fp32 out, LN is a separate cheap pass)
__global__ __launch_bounds__(128)
void k_oproj(const bf16* __restrict__ O, const bf16* __restrict__ Wo,
             const float* __restrict__ wob, const float* __restrict__ Ev,
             float* __restrict__ y) {
  int lane = threadIdx.x & 31, w = threadIdx.x >> 5;
  int row0 = (blockIdx.y << 6) + (w << 4), col0 = blockIdx.x << 6;
  v8f out[4];
  gemm_tile(O, Dn, Wo, Dn, row0, col0, Dn, lane, out);
  int half8 = ((lane >> 4) << 3), ln15 = lane & 15;
  for (int d = 0; d < 4; ++d)
    for (int e = 0; e < 8; ++e) {
      int m = row0 + e + half8, n = col0 + (d << 4) + ln15;
      y[(size_t)m * Dn + n] = out[d][e] + wob[n] + Ev[(size_t)m * Dn + n];
    }
}

// --------------------------- fused attention -------------------------------
// Grid: B*H * (L/64) = 512 WGs, 128 threads (4 waves). Wave w owns 16 query
// rows; stream 64-key blocks with online softmax. The rel-shift term is built
// per block by scatter of skewed S=qr.Qm^T WMMA tiles into an LDS T-block.
// k/v tiles are staged once per block into LDS and shared by all 4 waves.
#define STS 68   // padded fp32 row stride of the T block

__global__ __launch_bounds__(128)
void k_attn(const bf16* __restrict__ qk, const bf16* __restrict__ qr,
            const bf16* __restrict__ kmat, const bf16* __restrict__ vt,
            const bf16* __restrict__ qm, bf16* __restrict__ obf) {
  __shared__ __align__(16) bf16  sQk[64 * DKn];       // (q+cb) rows
  __shared__ __align__(16) bf16  sQr[80 * DKn];       // (q+pb) rows (+16 pad)
  __shared__ __align__(16) bf16  sK[64 * DKn];        // key tile (shared by waves)
  __shared__ __align__(16) bf16  sV[DKn * 64];        // v^T tile (dk-major)
  __shared__ __align__(16) float sT[64 * STS];        // shift(S) block
  __shared__ __align__(16) bf16  sP[4][16 * DKn];     // per-wave P restage

  const int tid  = threadIdx.x;
  const int lane = tid & 31;
  const int w    = tid >> 5;
  const int half8 = ((lane >> 4) << 3);
  const int ln15  = lane & 15;

  const int bh = blockIdx.x >> 5;           // L/64 = 32 row blocks per (b,h)
  const int i0 = (blockIdx.x & 31) << 6;
  const int h  = bh & (Hn - 1);

  const bf16* qkp = qk + (size_t)bh * Ln * DKn;
  const bf16* qrp = qr + (size_t)bh * Ln * DKn;
  const bf16* kp  = kmat + (size_t)bh * Ln * DKn;
  const bf16* vp  = vt + (size_t)bh * DKn * Ln;
  const bf16* qmp = qm + (size_t)h * Ln * DKn;

  // stage qk rows [i0,i0+64), qr rows [i0,i0+80) (clamped) into LDS
  for (int c = tid; c < 64 * (DKn / 8); c += 128) {
    int r = c >> 3, kc = (c & 7) << 3;
    *(uint4*)&sQk[r * DKn + kc] = *(const uint4*)&qkp[(size_t)(i0 + r) * DKn + kc];
  }
  for (int c = tid; c < 80 * (DKn / 8); c += 128) {
    int r = c >> 3, kc = (c & 7) << 3;
    int gr = i0 + r; if (gr > Ln - 1) gr = Ln - 1;
    *(uint4*)&sQr[r * DKn + kc] = *(const uint4*)&qrp[(size_t)gr * DKn + kc];
  }
  __syncthreads();

  float mrow[8], lrow[8];
  v8f acc[4];
  for (int e = 0; e < 8; ++e) { mrow[e] = -__builtin_inff(); lrow[e] = 0.f; }
  for (int d = 0; d < 4; ++d) acc[d] = zero8();

  for (int j0 = 0; j0 < Ln; j0 += 64) {
    // ---- stage k / v^T tiles (shared across waves), zero the T block ----
    for (int c = tid; c < 64 * (DKn / 8); c += 128) {
      int r = c >> 3, kc = (c & 7) << 3;
      *(uint4*)&sK[r * DKn + kc] = *(const uint4*)&kp[(size_t)(j0 + r) * DKn + kc];
      *(uint4*)&sV[r * 64 + kc]  = *(const uint4*)&vp[(size_t)r * Ln + j0 + kc];
    }
    for (int c = tid; c < 64 * STS; c += 128) sT[c] = 0.f;
    if (j0 + 64 < Ln) {       // hint next key/value block toward the caches
      __builtin_prefetch(kp + (size_t)(j0 + 64) * DKn + (tid << 4), 0, 1);
      __builtin_prefetch(vp + (size_t)(tid >> 1) * Ln + j0 + 64, 0, 1);
    }
    __syncthreads();

    const bool need1 = (j0 <= i0 + 63);       // lower/diag: j <= i branch
    const bool need2 = (j0 + 63 >= i0 + 2);   // upper/diag: j >= i+2 branch

    if (need1) {                              // wave w owns one 16-row tile
      int r0 = i0 + (w << 4);
      if (j0 <= r0 + 15) {
        int mlo = (Ln - 1) - (r0 + 15) + j0;
        int mhi = (Ln - 1) - r0 + j0 + 63;
        for (int mt = (mlo & ~15); mt <= mhi; mt += 16) {
          v8f s = zero8();
          for (int k0 = 0; k0 < DKn; k0 += 32) {
            v16bf a = frag_ld(sQr, DKn, r0 - i0, k0, lane);
            v16bf b = frag_ld_clamp(qmp, DKn, mt, k0, lane, Ln - 1);
            s = WMMA_BF16(a, b, s);
          }
          int m = mt + ln15;
          for (int e = 0; e < 8; ++e) {
            int i = r0 + e + half8;
            int j = m + i - (Ln - 1);         // m = L-1-(i-j)
            if (j >= j0 && j < j0 + 64 && j <= i)
              sT[(i - i0) * STS + (j - j0)] = s[e];
          }
        }
      }
    }
    if (need2) {                              // 5 row tiles over waves
      for (int t = w; t < 5; t += 4) {
        int r0 = i0 + (t << 4);
        if (r0 >= Ln) continue;
        int mhi = j0 + 62 - r0;
        if (mhi < 0) continue;
        int mlo = j0 - (r0 + 15) - 1;
        for (int mt = ((mlo < 0 ? 0 : mlo) & ~15); mt <= mhi; mt += 16) {
          v8f s = zero8();
          for (int k0 = 0; k0 < DKn; k0 += 32) {
            v16bf a = frag_ld(sQr, DKn, r0 - i0, k0, lane);
            v16bf b = frag_ld_clamp(qmp, DKn, mt, k0, lane, Ln - 1);
            s = WMMA_BF16(a, b, s);
          }
          int m = mt + ln15;
          for (int e = 0; e < 8; ++e) {
            int r = r0 + e + half8;           // S row r feeds A row r-1
            int i = r - 1;
            int j = m + r + 1;                // m = j - i - 2
            if (i >= i0 && i < i0 + 64 && j >= j0 && j < j0 + 64)
              sT[(i - i0) * STS + (j - j0)] = s[e];
          }
        }
      }
    }
    __syncthreads();

    // ---- scores: (q+cb) @ k^T + T   (k operands from LDS) ----
    v8f sc[4];
    {
      v16bf a0 = frag_ld(sQk, DKn, w << 4, 0, lane);
      v16bf a1 = frag_ld(sQk, DKn, w << 4, 32, lane);
      for (int ns = 0; ns < 4; ++ns) {
        v16bf b0 = frag_ld(sK, DKn, ns << 4, 0, lane);
        v16bf b1 = frag_ld(sK, DKn, ns << 4, 32, lane);
        v8f s = WMMA_BF16(a0, b0, zero8());
        s = WMMA_BF16(a1, b1, s);
        for (int e = 0; e < 8; ++e)
          s[e] += sT[((w << 4) + e + half8) * STS + (ns << 4) + ln15];
        sc[ns] = s;
      }
    }

    // ---- online softmax update (rows replicated over 16-lane halves) ----
    float bmax[8];
    for (int e = 0; e < 8; ++e)
      bmax[e] = fmaxf(fmaxf(sc[0][e], sc[1][e]), fmaxf(sc[2][e], sc[3][e]));
    for (int msk = 1; msk < 16; msk <<= 1)
      for (int e = 0; e < 8; ++e)
        bmax[e] = fmaxf(bmax[e], __shfl_xor(bmax[e], msk, 32));
    float scale[8];
    for (int e = 0; e < 8; ++e) {
      float nm = fmaxf(mrow[e], bmax[e]);
      scale[e] = exp2f((mrow[e] - nm) * LOG2E);
      mrow[e] = nm;
    }
    float rs[8];
    for (int e = 0; e < 8; ++e) rs[e] = 0.f;
    for (int ns = 0; ns < 4; ++ns)
      for (int e = 0; e < 8; ++e) {
        float p = exp2f((sc[ns][e] - mrow[e]) * LOG2E);
        sc[ns][e] = p;
        rs[e] += p;
      }
    for (int msk = 1; msk < 16; msk <<= 1)
      for (int e = 0; e < 8; ++e) rs[e] += __shfl_xor(rs[e], msk, 32);
    for (int e = 0; e < 8; ++e) lrow[e] = lrow[e] * scale[e] + rs[e];
    for (int d = 0; d < 4; ++d)
      for (int e = 0; e < 8; ++e) acc[d][e] *= scale[e];

    // ---- restage P (D-layout -> A-layout) and accumulate P @ v ----
    bf16* sPw = sP[w];
    for (int ns = 0; ns < 4; ++ns)
      for (int e = 0; e < 8; ++e)
        sPw[(e + half8) * DKn + (ns << 4) + ln15] = (bf16)sc[ns][e];
    asm volatile("s_wait_dscnt 0" ::: "memory");    // wave-local LDS RAW
    {
      v16bf pa0 = frag_ld(sPw, DKn, 0, 0, lane);
      v16bf pa1 = frag_ld(sPw, DKn, 0, 32, lane);
      for (int d = 0; d < 4; ++d) {
        v16bf b0 = frag_ld(sV, 64, d << 4, 0, lane);    // v^T: N=dk, K=pos
        v16bf b1 = frag_ld(sV, 64, d << 4, 32, lane);
        acc[d] = WMMA_BF16(pa0, b0, acc[d]);
        acc[d] = WMMA_BF16(pa1, b1, acc[d]);
      }
    }
    __syncthreads();   // protect sT/sK/sV reuse next iteration
  }

  // ---- normalize and store O (contiguous (B*H,L,DK) == reference reshape) ----
  for (int d = 0; d < 4; ++d)
    for (int e = 0; e < 8; ++e) {
      int row = i0 + (w << 4) + e + half8;
      int col = (d << 4) + ln15;
      obf[((size_t)bh * Ln + row) * DKn + col] = (bf16)(acc[d][e] / lrow[e]);
    }
}

// --------------------------- LayerNorm -------------------------------------
__global__ __launch_bounds__(256)
void k_ln(const float* __restrict__ y, const float* __restrict__ g,
          const float* __restrict__ bta, float* __restrict__ out) {
  int wave = (blockIdx.x * blockDim.x + threadIdx.x) >> 5;
  int lane = threadIdx.x & 31;
  if (wave >= BLn) return;
  const float* yr = y + (size_t)wave * Dn;
  float vals[16], s = 0.f, s2 = 0.f;
  for (int i = 0; i < 16; ++i) {
    float v = yr[lane + 32 * i];
    vals[i] = v; s += v; s2 += v * v;
  }
  for (int m = 1; m < 32; m <<= 1) {
    s  += __shfl_xor(s, m, 32);
    s2 += __shfl_xor(s2, m, 32);
  }
  float mu  = s * (1.f / Dn);
  float var = s2 * (1.f / Dn) - mu * mu;
  float inv = rsqrtf(var + 1e-5f);
  for (int i = 0; i < 16; ++i) {
    int c = lane + 32 * i;
    out[(size_t)wave * Dn + c] = (vals[i] - mu) * inv * g[c] + bta[c];
  }
}

// --------------------------- host orchestration ----------------------------
extern "C" void kernel_launch(void* const* d_in, const int* in_sizes, int n_in,
                              void* d_out, int out_size, void* d_ws, size_t ws_size,
                              hipStream_t stream) {
  (void)in_sizes; (void)n_in; (void)out_size; (void)ws_size;
  const float* E    = (const float*)d_in[0];
  const float* Ev   = (const float*)d_in[1];
  const float* R    = (const float*)d_in[2];
  const float* Wq   = (const float*)d_in[3];
  const float* Wke  = (const float*)d_in[4];
  const float* Wv   = (const float*)d_in[5];
  const float* Wkr  = (const float*)d_in[6];
  const float* cb   = (const float*)d_in[7];
  const float* pb   = (const float*)d_in[8];
  const float* Wo_w = (const float*)d_in[9];
  const float* Wo_b = (const float*)d_in[10];
  const float* ln_g = (const float*)d_in[11];
  const float* ln_b = (const float*)d_in[12];
  float* out = (float*)d_out;

  char* p = (char*)d_ws;
  auto carve = [&](size_t bytes) -> void* {
    void* r = (void*)p;
    p += (bytes + 255) & ~(size_t)255;
    return r;
  };
  bf16* E_bf   = (bf16*)carve((size_t)BLn * Dn * 2);
  bf16* Ev_bf  = (bf16*)carve((size_t)BLn * Dn * 2);
  bf16* RT_bf  = (bf16*)carve((size_t)Ln * Dn * 2);
  bf16* WqT    = (bf16*)carve((size_t)Dn * Dn * 2);
  bf16* WkeT   = (bf16*)carve((size_t)Dn * Dn * 2);
  bf16* WvT    = (bf16*)carve((size_t)Dn * Dn * 2);
  bf16* Wkr_bf = (bf16*)carve((size_t)Dn * Dn * 2);
  bf16* Wo_bf  = (bf16*)carve((size_t)Dn * Dn * 2);
  bf16* qk_bf  = (bf16*)carve((size_t)Bn * Hn * Ln * DKn * 2);
  bf16* qr_bf  = (bf16*)carve((size_t)Bn * Hn * Ln * DKn * 2);
  bf16* k_bf   = (bf16*)carve((size_t)Bn * Hn * Ln * DKn * 2);
  bf16* vT_bf  = (bf16*)carve((size_t)Bn * Hn * DKn * Ln * 2);
  bf16* Qm_bf  = (bf16*)carve((size_t)Hn * Ln * DKn * 2);
  bf16* O_bf   = (bf16*)carve((size_t)Bn * Hn * Ln * DKn * 2);
  float* ybuf  = (float*)carve((size_t)BLn * Dn * 4);

  const int nE = BLn * Dn;         // 2,097,152
  const int nW = Dn * Dn;          // 262,144
  const int nR = Dn * Ln;          // 1,048,576

  k_f2bf<<<dim3((nE + 255) / 256), dim3(256), 0, stream>>>(E, E_bf, nE);
  k_f2bf<<<dim3((nE + 255) / 256), dim3(256), 0, stream>>>(Ev, Ev_bf, nE);
  k_transpose_bf<<<dim3((nR + 255) / 256), dim3(256), 0, stream>>>(R, RT_bf, Dn, Ln);
  k_transpose_bf<<<dim3((nW + 255) / 256), dim3(256), 0, stream>>>(Wq, WqT, Dn, Dn);
  k_transpose_bf<<<dim3((nW + 255) / 256), dim3(256), 0, stream>>>(Wke, WkeT, Dn, Dn);
  k_transpose_bf<<<dim3((nW + 255) / 256), dim3(256), 0, stream>>>(Wv, WvT, Dn, Dn);
  k_f2bf<<<dim3((nW + 255) / 256), dim3(256), 0, stream>>>(Wkr, Wkr_bf, nW);
  k_f2bf<<<dim3((nW + 255) / 256), dim3(256), 0, stream>>>(Wo_w, Wo_bf, nW);

  k_proj_q<<<dim3(Dn / 64, BLn / 64), dim3(128), 0, stream>>>(E_bf, WqT, cb, pb, qk_bf, qr_bf);
  k_proj_k<<<dim3(Dn / 64, BLn / 64), dim3(128), 0, stream>>>(E_bf, WkeT, k_bf);
  k_proj_v<<<dim3(Dn / 64, BLn / 64), dim3(128), 0, stream>>>(Ev_bf, WvT, vT_bf);
  k_qm   <<<dim3(Dn / 64, Ln / 64),  dim3(128), 0, stream>>>(RT_bf, Wkr_bf, Qm_bf);

  k_attn<<<dim3(Bn * Hn * (Ln / 64)), dim3(128), 0, stream>>>(qk_bf, qr_bf, k_bf,
                                                              vT_bf, Qm_bf, O_bf);

  k_oproj<<<dim3(Dn / 64, BLn / 64), dim3(128), 0, stream>>>(O_bf, Wo_bf, Wo_b, Ev, ybuf);
  k_ln<<<dim3(BLn / 8), dim3(256), 0, stream>>>(ybuf, ln_g, ln_b, out);
}
